// BasicS2ConvV2_34351148434355
// MI455X (gfx1250) — compile-verified, matrix operand order checked by reading, not run on hardware
//
#include <hip/hip_runtime.h>
#include <hip/hip_bf16.h>
#include <stdint.h>

// ---- problem constants (from reference) ----
#define K_      13
#define A_      12
#define R_      12
#define NPARAM  36
#define DIN     32
#define DOUT    32
#define B_      2
#define P_      8192

#define KDIM  (DIN * K_ * A_)   // 4992 : GEMM K
#define NCOL  (DOUT * R_)       // 384  : GEMM N
#define KC    32                // K-chunk per step (one bf16 WMMA K)
#define NSTEP (KDIM / KC)       // 156
#define PTILE 64                // M rows per workgroup
#define NTILE 192               // N cols per workgroup
#define KP    34                // XS row stride (f32), padded vs 32 for banks
#define WP    40                // W LDS row stride (bf16), 80B = 16B aligned

typedef __attribute__((ext_vector_type(16))) __bf16 bf16x16;
typedef __attribute__((ext_vector_type(8)))  float  f32x8;

#if defined(__gfx1250__) && \
    __has_builtin(__builtin_amdgcn_global_load_async_to_lds_b32) && \
    __has_builtin(__builtin_amdgcn_global_load_async_to_lds_b128) && \
    __has_builtin(__builtin_amdgcn_s_wait_asynccnt)
#define USE_ASYNC 1
#pragma message("CDNA5: using global_load_async_to_lds + s_wait_asynccnt")
#else
#define USE_ASYNC 0
#pragma message("CDNA5: async-to-LDS builtins unavailable, sync LDS fallback")
#endif

// Types matching the builtin signatures revealed by the round-1 diagnostics:
//   b32 : param0 = AS1 (global) int*
//   b128: param0 = AS1 pointer to 16-byte int vector (vector_size(16))
typedef int vint4 __attribute__((vector_size(16)));
typedef __attribute__((address_space(1))) int   as1_int;
typedef __attribute__((address_space(3))) int   as3_int;
typedef __attribute__((address_space(1))) vint4 as1_int4;
typedef __attribute__((address_space(3))) vint4 as3_int4;

union Frag { unsigned u[8]; bf16x16 v; };

// ---------------------------------------------------------------------------
// Kernel 1: resolve gather W_eff and pre-split into bf16 hi/lo planes.
// Layout: whi/wlo[(n = d*R + r)][kk = (c*K + k)*A + a], row-major [NCOL][KDIM].
// ---------------------------------------------------------------------------
__global__ void weff_build_kernel(const float* __restrict__ W,
                                  const int* __restrict__ idx_map,
                                  const int* __restrict__ idxs_k,
                                  const int* __restrict__ idxs_a,
                                  unsigned short* __restrict__ whi,
                                  unsigned short* __restrict__ wlo) {
  int e = blockIdx.x * blockDim.x + threadIdx.x;
  if (e >= NCOL * KDIM) return;
  int n  = e / KDIM;
  int kk = e - n * KDIM;
  int c   = kk / (K_ * A_);
  int rem = kk - c * (K_ * A_);
  int k = rem / A_;
  int a = rem - k * A_;
  int d = n / R_;
  int r = n - d * R_;
  int a2 = idxs_a[(k * A_ + a)  * R_ + r];
  int k2 = idxs_k[(k * A_ + a2) * R_ + r];
  int m  = idx_map[k2 * A_ + a2];
  float w = W[(d * DIN + c) * NPARAM + m];
  unsigned u  = __builtin_bit_cast(unsigned, w);
  float    fh = __builtin_bit_cast(float, u & 0xFFFF0000u);
  float    fl = w - fh;
  whi[e] = (unsigned short)(u >> 16);
  wlo[e] = (unsigned short)(__builtin_bit_cast(unsigned, fl) >> 16);
}

// ---------------------------------------------------------------------------
// Kernel 2: batched GEMM OUT[p,(d,r)] += X[p,(c,k,a)] * W2[(c,k,a),(d,r)]
// using split-bf16 (hi/lo) -> 3x v_wmma_f32_16x16x32_bf16, f32 accumulate.
// Grid: (P/64, NCOL/192, B). Block: 256 threads = 8 waves (4 M x 2 Ngroups).
// ---------------------------------------------------------------------------
__launch_bounds__(256)
__global__ void s2conv_gemm_kernel(const float* __restrict__ x,
                                   const unsigned short* __restrict__ whi,
                                   const unsigned short* __restrict__ wlo,
                                   float* __restrict__ out) {
  __shared__ __align__(16) float          XS[2][PTILE][KP];  // fp32 x tile
  __shared__ __align__(16) unsigned short WH[2][NTILE][WP];  // bf16 hi W tile
  __shared__ __align__(16) unsigned short WL[2][NTILE][WP];  // bf16 lo W tile

  const int tid  = threadIdx.x;
  const int lane = tid & 31;
  const int w    = tid >> 5;
  const int mt   = w & 3;        // M-tile of this wave (0..3)
  const int ng   = w >> 2;       // N-group (0..1), 6 N-tiles each
  const int g    = lane >> 4;    // lane group (K-half select)
  const int ln   = lane & 15;    // row/col within tile
  const int p0   = blockIdx.x * PTILE;
  const int nh   = blockIdx.y;   // N half of 384 (0..1)
  const int b    = blockIdx.z;

  const float* xb = x + (size_t)b * DIN * K_ * P_ * A_;

  f32x8 acc[6] = {};

  auto issue = [&](int s, int bi) {
    // x tile: KC x PTILE fp32, stored [p][kk] so frag loads are b64-contiguous
#pragma unroll
    for (int j = 0; j < 8; ++j) {
      int idx = j * 256 + tid;             // 0..2047
      int kkl = idx & (KC - 1);
      int pl  = idx >> 5;
      unsigned kk  = (unsigned)(s * KC + kkl);
      unsigned c   = kk / (K_ * A_);
      unsigned rem = kk - c * (K_ * A_);
      unsigned k   = rem / A_;
      unsigned a   = rem - k * A_;
      int goff = (int)(((c * K_ + k) * P_ + (unsigned)(p0 + pl)) * A_ + a);
#if USE_ASYNC
      __builtin_amdgcn_global_load_async_to_lds_b32(
          (as1_int*)(xb + goff), (as3_int*)&XS[bi][pl][kkl], 0, 0);
#else
      XS[bi][pl][kkl] = xb[goff];
#endif
    }
    // W tiles: NTILE rows x KC bf16 (64B/row) as 4 x 16B chunks per row
#pragma unroll
    for (int j = 0; j < 3; ++j) {
      int idx = j * 256 + tid;             // 0..767
      int nl  = idx >> 2;
      int c16 = idx & 3;
      int off = (nh * NTILE + nl) * KDIM + s * KC + c16 * 8;
#if USE_ASYNC
      __builtin_amdgcn_global_load_async_to_lds_b128(
          (as1_int4*)(whi + off), (as3_int4*)&WH[bi][nl][c16 * 8], 0, 0);
      __builtin_amdgcn_global_load_async_to_lds_b128(
          (as1_int4*)(wlo + off), (as3_int4*)&WL[bi][nl][c16 * 8], 0, 0);
#else
      *(int4*)&WH[bi][nl][c16 * 8] = *(const int4*)(whi + off);
      *(int4*)&WL[bi][nl][c16 * 8] = *(const int4*)(wlo + off);
#endif
    }
  };

  issue(0, 0);

  for (int s = 0; s < NSTEP; ++s) {
    const int bi = s & 1;
#if USE_ASYNC
    __builtin_amdgcn_s_wait_asynccnt(0);
#endif
    __syncthreads();                 // tile s resident; all waves past step s-1
    if (s + 1 < NSTEP) issue(s + 1, bi ^ 1);   // prefetch next into other buf

    // ---- A fragments (x row, split to bf16 hi/lo) -------------------------
    // 16-bit A layout: lane m=ln, VGPR v packs K = {2v,2v+1} (+16 for v>=4), +8g
    Frag Ahi, Alo;
    {
      const float* xrow = &XS[bi][mt * 16 + ln][0];
#pragma unroll
      for (int v = 0; v < 8; ++v) {
        int kkA = (v < 4 ? 2 * v : 16 + 2 * (v - 4)) + 8 * g;
        float2 f = *(const float2*)(xrow + kkA);           // ds_load_b64
        unsigned u0 = __builtin_bit_cast(unsigned, f.x);
        unsigned u1 = __builtin_bit_cast(unsigned, f.y);
        float h0 = __builtin_bit_cast(float, u0 & 0xFFFF0000u);
        float h1 = __builtin_bit_cast(float, u1 & 0xFFFF0000u);
        float l0 = f.x - h0;
        float l1 = f.y - h1;
        Ahi.u[v] = (u1 & 0xFFFF0000u) | (u0 >> 16);
        Alo.u[v] = (__builtin_bit_cast(unsigned, l1) & 0xFFFF0000u) |
                   (__builtin_bit_cast(unsigned, l0) >> 16);
      }
    }

    // ---- 6 N-tiles: load B frags (pre-split bf16), 3 WMMAs each -----------
#pragma unroll
    for (int nt = 0; nt < 6; ++nt) {
      const int nloc = ng * 96 + nt * 16 + ln;
      const unsigned short* wrh = &WH[bi][nloc][0];
      const unsigned short* wrl = &WL[bi][nloc][0];
      Frag Bhi, Blo;
      const int o0 = 8 * g, o1 = 8 * g + 4, o2 = 16 + 8 * g, o3 = 20 + 8 * g;
      *(unsigned long long*)&Bhi.u[0] = *(const unsigned long long*)(wrh + o0);
      *(unsigned long long*)&Bhi.u[2] = *(const unsigned long long*)(wrh + o1);
      *(unsigned long long*)&Bhi.u[4] = *(const unsigned long long*)(wrh + o2);
      *(unsigned long long*)&Bhi.u[6] = *(const unsigned long long*)(wrh + o3);
      *(unsigned long long*)&Blo.u[0] = *(const unsigned long long*)(wrl + o0);
      *(unsigned long long*)&Blo.u[2] = *(const unsigned long long*)(wrl + o1);
      *(unsigned long long*)&Blo.u[4] = *(const unsigned long long*)(wrl + o2);
      *(unsigned long long*)&Blo.u[6] = *(const unsigned long long*)(wrl + o3);

      acc[nt] = __builtin_amdgcn_wmma_f32_16x16x32_bf16(
          false, Ahi.v, false, Bhi.v, (short)0, acc[nt], false, false);
      acc[nt] = __builtin_amdgcn_wmma_f32_16x16x32_bf16(
          false, Ahi.v, false, Blo.v, (short)0, acc[nt], false, false);
      acc[nt] = __builtin_amdgcn_wmma_f32_16x16x32_bf16(
          false, Alo.v, false, Bhi.v, (short)0, acc[nt], false, false);
    }
  }

  // ---- store: D VGPR i, lane -> (M = i + 8g, N = ln) ----------------------
#pragma unroll
  for (int nt = 0; nt < 6; ++nt) {
    int n = nh * NTILE + ng * 96 + nt * 16 + ln;
    int d = n / R_;
    int r = n - d * R_;
    float* ob = out + ((size_t)(b * DOUT + d) * P_ + (p0 + mt * 16 + g * 8)) * R_ + r;
#pragma unroll
    for (int i = 0; i < 8; ++i) ob[(size_t)i * R_] = acc[nt][i];
  }
}

// ---------------------------------------------------------------------------
extern "C" void kernel_launch(void* const* d_in, const int* in_sizes, int n_in,
                              void* d_out, int out_size, void* d_ws, size_t ws_size,
                              hipStream_t stream) {
  const float* x       = (const float*)d_in[0];
  const float* W       = (const float*)d_in[1];
  const int*   idx_map = (const int*)d_in[2];
  const int*   idxs_k  = (const int*)d_in[3];
  const int*   idxs_a  = (const int*)d_in[4];
  float*       out     = (float*)d_out;

  unsigned short* whi = (unsigned short*)d_ws;                 // [NCOL][KDIM] bf16
  unsigned short* wlo = whi + (size_t)NCOL * KDIM;             // [NCOL][KDIM] bf16

  const int total = NCOL * KDIM;
  weff_build_kernel<<<(total + 255) / 256, 256, 0, stream>>>(
      W, idx_map, idxs_k, idxs_a, whi, wlo);

  dim3 grid(P_ / PTILE, NCOL / NTILE, B_);  // (128, 2, 2)
  s2conv_gemm_kernel<<<grid, 256, 0, stream>>>(x, whi, wlo, out);
}
